// LSTMClassifier_66065186947527
// MI455X (gfx1250) — matrix operand order repaired and checked by reading, not compiled, not run on hardware
//
#include <hip/hip_runtime.h>
#include <hip/hip_bf16.h>
#include <stdint.h>

// LSTM classifier, CDNA5 (gfx1250, wave32, WMMA + async global->LDS staging).
// B=32, T=2048, V=32000, E=H=512, 4H=2048, C=4.

#define T_LEN 2048
#define B_SZ  32
#define E_DIM 512
#define H_DIM 512
#define Z_DIM 2048   // 4H
#define NWG   16     // persistent workgroups in the scan kernel

typedef __attribute__((ext_vector_type(16))) __bf16 v16bf;
typedef __attribute__((ext_vector_type(8)))  float  v8f;

__device__ __forceinline__ unsigned short f2bf(float f) {
    unsigned u = __float_as_uint(f);
    u += 0x7fffu + ((u >> 16) & 1u);      // round-to-nearest-even
    return (unsigned short)(u >> 16);
}
__device__ __forceinline__ float sigmf(float x) { return 1.0f / (1.0f + __expf(-x)); }

// ---- CDNA5 async global->LDS copy (ASYNCcnt path, §15.18.3 op 98) ----------
// Copies 32 KB: 256 threads x 16 B x 8 iterations. lds_dst is an LDS byte
// address (low 32 bits of the flat shared pointer), gsrc a global pointer.
__device__ __forceinline__ void async_copy32k(unsigned lds_dst, const void* gsrc,
                                              int tid) {
    const char* g = (const char*)gsrc;
#pragma unroll
    for (int i = 0; i < 8; i++) {
        const unsigned off = (unsigned)(tid * 16 + i * 4096);
        asm volatile("global_load_async_to_lds_b128 %0, %1, off"
                     :: "v"(lds_dst + off), "v"(g + off)
                     : "memory");
    }
}
__device__ __forceinline__ void wait_async() {
    asm volatile("s_wait_asynccnt 0x0" ::: "memory");
}

// ---------------------------------------------------------------------------
// Pack a [512 x 2048] fp32 weight matrix into bf16 WMMA B-tiles (32x16 each).
// B-tile lane layout (ISA 7.12.2): col = lane&15, K = (lane>>4)*8 + (j&7) + ((j>>3)<<4).
// Tile order: (kt*128 + nt), 512 bf16 per tile, lane-contiguous 16 bf16.
// ---------------------------------------------------------------------------
__global__ __launch_bounds__(256) void pack_w_kernel(const float* __restrict__ W,
                                                     unsigned short* __restrict__ out) {
    const int wid  = threadIdx.x >> 5, lane = threadIdx.x & 31;
    const int tile = blockIdx.x * 8 + wid;            // 2048 tiles = 16 kt * 128 nt
    const int kt = tile >> 7, nt = tile & 127;
    const int n  = (nt << 4) + (lane & 15);
    const int kb = (kt << 5) + ((lane >> 4) << 3);
    unsigned short* dst = out + ((size_t)tile << 9) + (lane << 4);
#pragma unroll
    for (int j = 0; j < 16; j++)
        dst[j] = f2bf(W[(size_t)(kb + (j & 7) + ((j >> 3) << 4)) * Z_DIM + n]);
}

// ---------------------------------------------------------------------------
// Gather embeddings for every (t, b) and pack into bf16 WMMA A-tiles (16x32).
// A-tile lane layout (ISA 7.12.2, 16-bit A 16x32):
//   row M = lane&15;  bf16 j in lane: k = (j&7) + ((lane>>4)&1)*8 + ((j>>3)<<4)
// Tile id = t*32 + mt*16 + kt, 512 bf16/tile, lane-contiguous.
// ---------------------------------------------------------------------------
__global__ __launch_bounds__(256) void pack_e_kernel(const int* __restrict__ x,
                                                     const float* __restrict__ emb,
                                                     unsigned short* __restrict__ out) {
    const int wid  = threadIdx.x >> 5, lane = threadIdx.x & 31;
    const int tile = blockIdx.x * 8 + wid;            // 65536 tiles
    const int kt = tile & 15, mt = (tile >> 4) & 1, t = tile >> 5;
    const int b  = (mt << 4) + (lane & 15);
    const int tok = x[b * T_LEN + t];
    const float* er = emb + (size_t)tok * E_DIM + (kt << 5);
    unsigned short* dst = out + ((size_t)tile << 9) + (lane << 4);
    const int kh = (lane >> 4) & 1;
#pragma unroll
    for (int j = 0; j < 16; j++) {
        int k = (j & 7) + kh * 8 + ((j >> 3) << 4);
        dst[j] = f2bf(er[k]);
    }
}

// ---------------------------------------------------------------------------
// Per-batch valid lengths: clamp(sum(attn[b,:]), 1, T)
// ---------------------------------------------------------------------------
__global__ __launch_bounds__(256) void lengths_kernel(const int* __restrict__ attn,
                                                      int* __restrict__ len) {
    __shared__ int red[256];
    const int b = blockIdx.x;
    int s = 0;
    for (int i = threadIdx.x; i < T_LEN; i += 256) s += attn[b * T_LEN + i];
    red[threadIdx.x] = s;
    __syncthreads();
    for (int st = 128; st > 0; st >>= 1) {
        if (threadIdx.x < st) red[threadIdx.x] += red[threadIdx.x + st];
        __syncthreads();
    }
    if (threadIdx.x == 0) {
        int L = red[0];
        len[b] = L < 1 ? 1 : (L > T_LEN ? T_LEN : L);
    }
}

__global__ void init_kernel(unsigned* cnt) { if (threadIdx.x == 0) *cnt = 0u; }

// ---------------------------------------------------------------------------
// Persistent 2-layer LSTM scan. 16 WGs x 256 threads (8 wave32s).
// WG wg owns hidden columns [wg*32, wg*32+32) of each layer
//   -> z columns {g*512 + wg*32 + [0,32)} for gates g = i,f,g,o  (8 N-tiles)
// Wave wid: gate = wid>>1, 16-col sub-tile = wid&1.
// Per step & layer:
//   1) async-stage both A operand blocks (32 KB each) into LDS once per WG
//   2) 2 M-tiles x 16 k-steps x 2 fused GEMMs = 64 WMMAs/wave, A from LDS,
//      B streamed from L2-resident packed weights
// h state republished to global each step in bf16 A-packed layout (kt = wg),
// double-buffered on step parity; c state lives in LDS.
// ---------------------------------------------------------------------------
__global__ __launch_bounds__(256, 1) void lstm_scan_kernel(
    const unsigned short* __restrict__ epack,
    const unsigned short* __restrict__ wx0p, const unsigned short* __restrict__ wh0p,
    const unsigned short* __restrict__ wx1p, const unsigned short* __restrict__ wh1p,
    const float* __restrict__ b0, const float* __restrict__ b1,
    const int* __restrict__ len,
    unsigned short* __restrict__ h0p, unsigned short* __restrict__ h1p,
    float* __restrict__ hsel, unsigned* __restrict__ cnt) {

    __shared__ __align__(1024) unsigned char ldsA[65536]; // [0,32K)=A0, [32K,64K)=A1
    __shared__ float zbuf[4][B_SZ][32];   // [gate][batch][local h-col]
    __shared__ float cst[2][B_SZ][32];    // cell state, layers 0/1
    __shared__ int   lenS[B_SZ];

    const int tid  = threadIdx.x, wid = tid >> 5, lane = tid & 31;
    const int wg   = blockIdx.x;
    const int gate = wid >> 1, sub = wid & 1, nloc = lane & 15;
    const int zcol = gate * H_DIM + wg * 32 + sub * 16 + nloc;  // global z column
    const int nt   = zcol >> 4;                                 // N-tile in packed W
    const float bias0 = b0[zcol], bias1 = b1[zcol];
    const unsigned ldsA_base = (unsigned)(uintptr_t)(&ldsA[0]); // LDS byte address

    for (int i = tid; i < B_SZ * 32; i += 256) {
        ((float*)cst)[i] = 0.f;
        ((float*)cst)[B_SZ * 32 + i] = 0.f;
    }
    if (tid < B_SZ) lenS[tid] = len[tid];
    // zero this WG's h-pack slices (kt = wg), both parities & M-tiles
    for (int p = 0; p < 2; p++)
        for (int mt = 0; mt < 2; mt++) {
            unsigned short* d0 = h0p + (((p * 2 + mt) * 16 + wg) << 9);
            unsigned short* d1 = h1p + (((p * 2 + mt) * 16 + wg) << 9);
            for (int i = tid; i < 512; i += 256) { d0[i] = 0; d1[i] = 0; }
        }

    unsigned gen = 0;
#define GRID_BARRIER()                                                         \
    do {                                                                       \
        gen++;                                                                 \
        __syncthreads();                                                       \
        if (tid == 0) {                                                        \
            __threadfence();                                                   \
            atomicAdd(cnt, 1u);                                                \
            const unsigned tgt = gen * NWG;                                    \
            while (*((volatile unsigned*)cnt) < tgt)                           \
                __builtin_amdgcn_s_sleep(2);                                   \
            __threadfence();                                                   \
        }                                                                      \
        __syncthreads();                                                       \
    } while (0)

    GRID_BARRIER();   // initial state visible everywhere

    for (int t = 0; t < T_LEN; t++) {
        const int wp = t & 1, rp = wp ^ 1;

        // ---- stage layer-0 A operands: e(t) and h0(t-1), 32 KB each ----
        async_copy32k(ldsA_base,         epack + ((size_t)t << 14), tid);
        async_copy32k(ldsA_base + 32768, h0p + (rp << 14), tid);
        wait_async();
        __syncthreads();

        // ----- layer 0: z0 = e_t@Wx0 + h0@Wh0 + b0 -----
        for (int mt = 0; mt < 2; mt++) {
            v8f acc;
#pragma unroll
            for (int r = 0; r < 8; r++) acc[r] = bias0;
            const unsigned char* aBase = ldsA + ((mt * 16) << 10) + (lane << 5);
            const unsigned short* bx = wx0p + ((size_t)nt << 9) + (lane << 4);
            const unsigned short* bh = wh0p + ((size_t)nt << 9) + (lane << 4);
#pragma unroll 4
            for (int kt = 0; kt < 16; kt++) {
                v16bf aE = *(const v16bf*)(aBase + (kt << 10));
                v16bf bX = *(const v16bf*)(bx + ((size_t)kt << 16));
                acc = __builtin_amdgcn_wmma_f32_16x16x32_bf16(false, aE, false, bX,
                                                              (short)0, acc, false, false);
                v16bf aH = *(const v16bf*)(aBase + 32768 + (kt << 10));
                v16bf bH = *(const v16bf*)(bh + ((size_t)kt << 16));
                acc = __builtin_amdgcn_wmma_f32_16x16x32_bf16(false, aH, false, bH,
                                                              (short)0, acc, false, false);
            }
            const int mb = (mt << 4) + ((lane >> 4) << 3);   // C layout: M = r (+8 hi lanes)
#pragma unroll
            for (int r = 0; r < 8; r++) zbuf[gate][mb + r][(sub << 4) + nloc] = acc[r];
        }
        __syncthreads();
        // gates -> c0, h0; republish h0 in bf16 A-packed layout (parity wp, kt = wg)
        for (int i = tid; i < B_SZ * 32; i += 256) {
            const int b = i >> 5, j = i & 31;
            float c = sigmf(zbuf[1][b][j]) * cst[0][b][j]
                    + sigmf(zbuf[0][b][j]) * tanhf(zbuf[2][b][j]);
            cst[0][b][j] = c;
            float h = sigmf(zbuf[3][b][j]) * tanhf(c);
            const int lane2 = (b & 15) + (((j >> 3) & 1) << 4);
            const int jj    = (j & 7) | (((j >> 4) & 1) << 3);
            h0p[(((wp * 2 + (b >> 4)) * 16 + wg) << 9) + (lane2 << 4) + jj] = f2bf(h);
        }
        GRID_BARRIER();   // all h0(t) slices visible

        // ---- stage layer-1 A operands: h0(t) and h1(t-1), 32 KB each ----
        async_copy32k(ldsA_base,         h0p + (wp << 14), tid);
        async_copy32k(ldsA_base + 32768, h1p + (rp << 14), tid);
        wait_async();
        __syncthreads();

        // ----- layer 1: z1 = h0@Wx1 + h1@Wh1 + b1 -----
        for (int mt = 0; mt < 2; mt++) {
            v8f acc;
#pragma unroll
            for (int r = 0; r < 8; r++) acc[r] = bias1;
            const unsigned char* aBase = ldsA + ((mt * 16) << 10) + (lane << 5);
            const unsigned short* bx = wx1p + ((size_t)nt << 9) + (lane << 4);
            const unsigned short* bh = wh1p + ((size_t)nt << 9) + (lane << 4);
#pragma unroll 4
            for (int kt = 0; kt < 16; kt++) {
                v16bf aA = *(const v16bf*)(aBase + (kt << 10));
                v16bf bX = *(const v16bf*)(bx + ((size_t)kt << 16));
                acc = __builtin_amdgcn_wmma_f32_16x16x32_bf16(false, aA, false, bX,
                                                              (short)0, acc, false, false);
                v16bf aB = *(const v16bf*)(aBase + 32768 + (kt << 10));
                v16bf bH = *(const v16bf*)(bh + ((size_t)kt << 16));
                acc = __builtin_amdgcn_wmma_f32_16x16x32_bf16(false, aB, false, bH,
                                                              (short)0, acc, false, false);
            }
            const int mb = (mt << 4) + ((lane >> 4) << 3);
#pragma unroll
            for (int r = 0; r < 8; r++) zbuf[gate][mb + r][(sub << 4) + nloc] = acc[r];
        }
        __syncthreads();
        for (int i = tid; i < B_SZ * 32; i += 256) {
            const int b = i >> 5, j = i & 31;
            float c = sigmf(zbuf[1][b][j]) * cst[1][b][j]
                    + sigmf(zbuf[0][b][j]) * tanhf(zbuf[2][b][j]);
            cst[1][b][j] = c;
            float h = sigmf(zbuf[3][b][j]) * tanhf(c);
            const int lane2 = (b & 15) + (((j >> 3) & 1) << 4);
            const int jj    = (j & 7) | (((j >> 4) & 1) << 3);
            h1p[(((wp * 2 + (b >> 4)) * 16 + wg) << 9) + (lane2 << 4) + jj] = f2bf(h);
            if (lenS[b] - 1 == t) hsel[b * H_DIM + wg * 32 + j] = h;
        }
        GRID_BARRIER();   // all h1(t) slices visible
    }
#undef GRID_BARRIER
}

// ---------------------------------------------------------------------------
// Head: out[b,c] = hsel[b,:] @ fcW[:,c] + fcb[c]   (32x512x4, fp32)
// ---------------------------------------------------------------------------
__global__ __launch_bounds__(128) void head_kernel(const float* __restrict__ hsel,
                                                   const float* __restrict__ fcW,
                                                   const float* __restrict__ fcb,
                                                   float* __restrict__ out) {
    const int b = threadIdx.x >> 2, c = threadIdx.x & 3;
    float s = fcb[c];
    for (int k = 0; k < H_DIM; k++) s += hsel[b * H_DIM + k] * fcW[k * 4 + c];
    out[b * 4 + c] = s;
}

extern "C" void kernel_launch(void* const* d_in, const int* in_sizes, int n_in,
                              void* d_out, int out_size, void* d_ws, size_t ws_size,
                              hipStream_t stream) {
    (void)in_sizes; (void)n_in; (void)out_size; (void)ws_size;
    const int*   x    = (const int*)d_in[0];
    const int*   attn = (const int*)d_in[1];
    const float* emb  = (const float*)d_in[2];
    const float* Wx0  = (const float*)d_in[3];
    const float* Wh0  = (const float*)d_in[4];
    const float* b0   = (const float*)d_in[5];
    const float* Wx1  = (const float*)d_in[6];
    const float* Wh1  = (const float*)d_in[7];
    const float* b1   = (const float*)d_in[8];
    const float* fcW  = (const float*)d_in[9];
    const float* fcb  = (const float*)d_in[10];

    // workspace layout (bytes)
    char* ws = (char*)d_ws;
    unsigned*       cnt   = (unsigned*)(ws + 0);           //   256 B
    int*            len   = (int*)(ws + 256);              //   256 B
    float*          hsel  = (float*)(ws + 4096);           //  64 KB
    unsigned short* h0p   = (unsigned short*)(ws + 69632); //  64 KB (2 parity x 32 tiles)
    unsigned short* h1p   = (unsigned short*)(ws + 135168);//  64 KB
    unsigned short* wx0p  = (unsigned short*)(ws + 262144);          // 2 MB each
    unsigned short* wh0p  = (unsigned short*)(ws + 262144 + (1u<<21));
    unsigned short* wx1p  = (unsigned short*)(ws + 262144 + (2u<<21));
    unsigned short* wh1p  = (unsigned short*)(ws + 262144 + (3u<<21));
    unsigned short* epack = (unsigned short*)(ws + 262144 + (4u<<21)); // 64 MB

    init_kernel<<<1, 64, 0, stream>>>(cnt);
    pack_w_kernel<<<256, 256, 0, stream>>>(Wx0, wx0p);
    pack_w_kernel<<<256, 256, 0, stream>>>(Wh0, wh0p);
    pack_w_kernel<<<256, 256, 0, stream>>>(Wx1, wx1p);
    pack_w_kernel<<<256, 256, 0, stream>>>(Wh1, wh1p);
    pack_e_kernel<<<8192, 256, 0, stream>>>(x, emb, epack);
    lengths_kernel<<<32, 256, 0, stream>>>(attn, len);
    lstm_scan_kernel<<<NWG, 256, 0, stream>>>(epack, wx0p, wh0p, wx1p, wh1p,
                                              b0, b1, len, h0p, h1p, hsel, cnt);
    head_kernel<<<1, 128, 0, stream>>>(hsel, fcW, fcb, (float*)d_out);
}
